// QCNNFeatureExtractorQuantum_65481071405481
// MI455X (gfx1250) — matrix operand order repaired and checked by reading, not compiled, not run on hardware
//
#include <hip/hip_runtime.h>
#include <random>
#include <cstdint>

// ---------------------------------------------------------------------------
// QCNN quantum feature extractor, restructured for CDNA5 (gfx1250) WMMA.
//
//   state_final = U * Psi[b]      (U: fixed 256x256 complex unitary,
//                                  Psi[b]: real product state from encoding)
//   out[b,q]    = sum_j sign(q,j) * |S[b,j]|^2,  S = Psi @ U^T
//
// Kernel 1 builds U (per-column, in LDS) and stores it as f16 hi/lo pairs in
// WMMA B-fragment layout.  Kernel 2 does the 32768x512x256 split-f16 GEMM
// with v_wmma_f32_16x16x32_f16 (3-product Markidis split -> ~fp32 accuracy),
// fused with |.|^2 and the sign-matrix reduction.
// ---------------------------------------------------------------------------

#define N_QUBITS 8
#define QDIM     256          // 2^8
#define N_OPS    20
#define N_GATES  36           // 20 RL ops + 8 * (RX, RY)

typedef __attribute__((ext_vector_type(16))) _Float16 v16h;
typedef __attribute__((ext_vector_type(8)))  float    v8f;

struct OpPack { int op[N_GATES]; };  // kind[1:0] | w0[4:2] | w1[7:5] | angle_idx[15:8]

// B-fragment storage layout (half-elements), contiguous k order (A uses the same):
//   lane = (N & 15) + 16*(K32 >> 4),  half index h = K32 & 15
__host__ __device__ inline int bfrag_off(int c, int part, int t, int kc, int lane, int h) {
    return (((((c * 2 + part) * 16 + t) * 8 + kc) * 32 + lane) * 16) + h;
}

// ---------------------------------------------------------------------------
// Kernel 1: build U = G36 ... G1 column-by-column; one block per column.
// Gather formulation per gate; double-buffered state in LDS.
// ---------------------------------------------------------------------------
__global__ __launch_bounds__(256)
void build_unitary_kernel(const float* __restrict__ rl,
                          const float* __restrict__ trx,
                          const float* __restrict__ trya,
                          OpPack P, _Float16* __restrict__ bfrag) {
    __shared__ float sre[2][QDIM];
    __shared__ float sim[2][QDIM];
    const int idx = threadIdx.x;     // basis index (row of U)
    const int col = blockIdx.x;      // column of U  == K index of B

    sre[0][idx] = (idx == col) ? 1.0f : 0.0f;
    sim[0][idx] = 0.0f;
    __syncthreads();

    int cur = 0;
    for (int g = 0; g < N_GATES; ++g) {
        const int op   = P.op[g];
        const int kind = op & 3;
        const int w0   = (op >> 2) & 7;
        const int w1   = (op >> 5) & 7;
        const int ai   = op >> 8;
        const int nxt  = cur ^ 1;

        if (kind == 3) {                      // CNOT(control=w0, target=w1)
            const int cm = 1 << (7 - w0);
            const int tm = 1 << (7 - w1);
            const int src = (idx & cm) ? (idx ^ tm) : idx;
            sre[nxt][idx] = sre[cur][src];
            sim[nxt][idx] = sim[cur][src];
        } else {
            const float ang = 0.5f * (ai < 20 ? rl[ai] : (ai == 20 ? trx[0] : trya[0]));
            const float c = cosf(ang), s = sinf(ang);
            const int p = 7 - w0;
            const int partner = idx ^ (1 << p);
            const int b = (idx >> p) & 1;
            const float r0 = sre[cur][idx],     i0 = sim[cur][idx];
            const float r1 = sre[cur][partner], i1 = sim[cur][partner];
            float nr, ni;
            if (kind == 0) {                  // RX: new = c*self - i*s*partner
                nr = c * r0 + s * i1;
                ni = c * i0 - s * r1;
            } else if (kind == 1) {           // RY
                const float ss = b ? s : -s;
                nr = c * r0 + ss * r1;
                ni = c * i0 + ss * i1;
            } else {                          // RZ: (c -/+ i*s) * self
                const float ss = b ? -s : s;
                nr = c * r0 + ss * i0;
                ni = c * i0 - ss * r0;
            }
            sre[nxt][idx] = nr;
            sim[nxt][idx] = ni;
        }
        __syncthreads();
        cur ^= 1;
    }

    // B[K=col][N=idx] = U[idx, col]; scatter into WMMA fragment layout.
    const int i  = idx;
    const int t  = i >> 4, n = i & 15;
    const int kc = col >> 5, kk = col & 31;
    const int lane = n + ((kk >> 4) << 4);
    const int h    = kk & 15;
    #pragma unroll
    for (int c = 0; c < 2; ++c) {
        const float v = c ? sim[cur][idx] : sre[cur][idx];
        const _Float16 hi = (_Float16)v;
        const _Float16 lo = (_Float16)(v - (float)hi);
        bfrag[bfrag_off(c, 0, t, kc, lane, h)] = hi;
        bfrag[bfrag_off(c, 1, t, kc, lane, h)] = lo;
    }
}

// ---------------------------------------------------------------------------
// Kernel 2: fused GEMM + |.|^2 + sign reduction.  8 waves/block, 16 rows/wave.
// ---------------------------------------------------------------------------
__global__ __launch_bounds__(256)
void qcnn_probs_wmma_kernel(const float* __restrict__ x,
                            const _Float16* __restrict__ bfrag,
                            float* __restrict__ out) {
    const int lane  = threadIdx.x & 31;
    const int wave  = threadIdx.x >> 5;
    const int rowbase = (blockIdx.x * 8 + wave) * 16;
    const int m     = lane & 15;
    const int khalf = lane >> 4;

    // --- encoding trig for row rowbase+m ---
    float cw[N_QUBITS], sw[N_QUBITS];
    const float* xr = x + (size_t)(rowbase + m) * N_QUBITS;
    #pragma unroll
    for (int w = 0; w < N_QUBITS; ++w) {
        const float ha = 0.5f * xr[w];
        cw[w] = cosf(ha);
        sw[w] = sinf(ha);
    }
    // nibble product tables: Psi(k) = ph[k>>4] * pl[k&15]
    float ph[16], pl[16];
    #pragma unroll
    for (int tn = 0; tn < 16; ++tn) {
        ph[tn] = ((tn & 8) ? sw[0] : cw[0]) * ((tn & 4) ? sw[1] : cw[1]) *
                 ((tn & 2) ? sw[2] : cw[2]) * ((tn & 1) ? sw[3] : cw[3]);
        pl[tn] = ((tn & 8) ? sw[4] : cw[4]) * ((tn & 4) ? sw[5] : cw[5]) *
                 ((tn & 2) ? sw[6] : cw[6]) * ((tn & 1) ? sw[7] : cw[7]);
    }

    // --- build A fragments (contiguous k order, as B), f16 hi/lo split ---
    v16h a_hi[8], a_lo[8];
    #pragma unroll
    for (int kc = 0; kc < 8; ++kc) {
        #pragma unroll
        for (int j = 0; j < 16; ++j) {
            // global K = kc*32 + 16*khalf + j
            const int jj = j & 7;
            const float phi = ph[2 * kc + khalf];
            const float plo = pl[j];
            const float psi = phi * plo;
            const _Float16 hi = (_Float16)psi;
            a_hi[kc][j] = hi;
            a_lo[kc][j] = (_Float16)(psi - (float)hi);
        }
    }

    // --- GEMM over 16 N-tiles of 16 output-state indices, fused reduce ---
    float sacc[8][8];
    #pragma unroll
    for (int vr = 0; vr < 8; ++vr)
        #pragma unroll
        for (int q = 0; q < 8; ++q) sacc[vr][q] = 0.0f;

    for (int t = 0; t < 16; ++t) {
        v8f acc_re = {}, acc_im = {};
        #pragma unroll
        for (int kc = 0; kc < 8; ++kc) {
            const v16h brh = *(const v16h*)(bfrag + bfrag_off(0, 0, t, kc, lane, 0));
            const v16h brl = *(const v16h*)(bfrag + bfrag_off(0, 1, t, kc, lane, 0));
            const v16h bih = *(const v16h*)(bfrag + bfrag_off(1, 0, t, kc, lane, 0));
            const v16h bil = *(const v16h*)(bfrag + bfrag_off(1, 1, t, kc, lane, 0));
            // Markidis split: hi*hi + hi*lo + lo*hi, f32 accumulate
            acc_re = __builtin_amdgcn_wmma_f32_16x16x32_f16(false, a_hi[kc], false, brh, (short)0, acc_re, false, false);
            acc_re = __builtin_amdgcn_wmma_f32_16x16x32_f16(false, a_hi[kc], false, brl, (short)0, acc_re, false, false);
            acc_re = __builtin_amdgcn_wmma_f32_16x16x32_f16(false, a_lo[kc], false, brh, (short)0, acc_re, false, false);
            acc_im = __builtin_amdgcn_wmma_f32_16x16x32_f16(false, a_hi[kc], false, bih, (short)0, acc_im, false, false);
            acc_im = __builtin_amdgcn_wmma_f32_16x16x32_f16(false, a_hi[kc], false, bil, (short)0, acc_im, false, false);
            acc_im = __builtin_amdgcn_wmma_f32_16x16x32_f16(false, a_lo[kc], false, bih, (short)0, acc_im, false, false);
        }
        // this lane holds D rows m = vr + 8*khalf, column j = t*16 + (lane&15)
        const int j = t * 16 + m;
        #pragma unroll
        for (int vr = 0; vr < 8; ++vr) {
            const float re = acc_re[vr], im = acc_im[vr];
            const float prob = re * re + im * im;
            #pragma unroll
            for (int q = 0; q < 8; ++q)
                sacc[vr][q] += ((j >> (7 - q)) & 1) ? -prob : prob;
        }
    }

    // --- reduce across the 16 lanes of each half (xor masks stay in-half) ---
    #pragma unroll
    for (int vr = 0; vr < 8; ++vr) {
        #pragma unroll
        for (int q = 0; q < 8; ++q) {
            float v = sacc[vr][q];
            v += __shfl_xor(v, 1, 32);
            v += __shfl_xor(v, 2, 32);
            v += __shfl_xor(v, 4, 32);
            v += __shfl_xor(v, 8, 32);
            sacc[vr][q] = v;
        }
    }
    if (m == 0) {  // lanes 0 (rows +0..7) and 16 (rows +8..15)
        #pragma unroll
        for (int vr = 0; vr < 8; ++vr) {
            const int b = rowbase + vr + 8 * khalf;
            #pragma unroll
            for (int q = 0; q < 8; ++q)
                out[(size_t)b * 8 + q] = sacc[vr][q];
        }
    }
}

// ---------------------------------------------------------------------------
// Host: reproduce _RL_OPS from np.random.RandomState(0).
//   - raw stream: MT19937 seeded 0 (std::mt19937 matches numpy init/temper)
//   - randint(n) [dtype int64]: next64 = (w1<<32)|w2, masked  -> w2 & (n-1)
//   - choice(8,2,replace=False) = permutation(8)[:2]; shuffle uses masked
//     rejection on single 32-bit draws (numpy random_interval, max<=2^32)
// ---------------------------------------------------------------------------
static void build_ops(OpPack& P) {
    std::mt19937 gen(0);
    auto nx = [&]() -> uint32_t { return (uint32_t)gen(); };
    auto randint_pow2 = [&](uint32_t n) -> uint32_t {
        (void)nx();                 // high 32 bits of the 64-bit draw
        return nx() & (n - 1);      // low bits, mask == n-1 (no rejection)
    };
    int g = 0;
    for (int k = 0; k < N_OPS; ++k) {
        const uint32_t kind = randint_pow2(4);   // 0=rx 1=ry 2=rz 3=cnot
        if (kind == 3) {
            int a[8] = {0, 1, 2, 3, 4, 5, 6, 7};
            for (int i = 7; i >= 1; --i) {
                uint32_t mask = 1;
                while (mask < (uint32_t)i) mask = (mask << 1) | 1;
                uint32_t j;
                do { j = nx() & mask; } while (j > (uint32_t)i);
                const int tv = a[i]; a[i] = a[(int)j]; a[(int)j] = tv;
            }
            P.op[g++] = 3 | (a[0] << 2) | (a[1] << 5);
        } else {
            const uint32_t w = randint_pow2(8);
            P.op[g++] = (int)kind | ((int)w << 2) | (k << 8);
        }
    }
    for (int w = 0; w < N_QUBITS; ++w) {
        P.op[g++] = 0 | (w << 2) | (20 << 8);   // RX(theta_rx/2)
        P.op[g++] = 1 | (w << 2) | (21 << 8);   // RY(theta_ry/2)
    }
}

extern "C" void kernel_launch(void* const* d_in, const int* in_sizes, int n_in,
                              void* d_out, int out_size, void* d_ws, size_t ws_size,
                              hipStream_t stream) {
    (void)n_in; (void)out_size; (void)ws_size;
    OpPack P;
    build_ops(P);

    const float* x    = (const float*)d_in[0];   // [BSZ, 8]
    const float* rl   = (const float*)d_in[1];   // [20]
    const float* trx  = (const float*)d_in[2];   // [1]
    const float* trya = (const float*)d_in[3];   // [1]
    _Float16*    bfrag = (_Float16*)d_ws;        // 512 KB B-fragment store

    const int bsz = in_sizes[0] / N_QUBITS;

    build_unitary_kernel<<<QDIM, 256, 0, stream>>>(rl, trx, trya, P, bfrag);
    qcnn_probs_wmma_kernel<<<bsz / 128, 256, 0, stream>>>(x, bfrag, (float*)d_out);
}
